// MultiHeadAttention_85538568667794
// MI455X (gfx1250) — matrix operand order repaired and checked by reading, not compiled, not run on hardware
//
#include <hip/hip_runtime.h>

// ---------------------------------------------------------------------------
// CDNA5 (gfx1250) DeBERTa-style banded attention, bf16 WMMA everywhere.
// B=4, T=1024, C=1024, H=16, hs=64, WINDOW=256.
// ---------------------------------------------------------------------------

typedef __attribute__((ext_vector_type(16))) __bf16 v16bf;
typedef __attribute__((ext_vector_type(8)))  float  v8f;

union BF16x16 { unsigned short u[16]; v16bf v; };

#define WMMA_BF16(a, b, c) \
  __builtin_amdgcn_wmma_f32_16x16x32_bf16(false, (a), false, (b), (short)0, (c), false, false)

__device__ __forceinline__ unsigned short f2bf(float f) {
  union { float f; unsigned int u; } x; x.f = f;
  unsigned int u = x.u;
  unsigned int r = u + 0x7FFFu + ((u >> 16) & 1u);   // round-to-nearest-even
  return (unsigned short)(r >> 16);
}

// A fragment (16x32, bf16) from LDS: lane L<16 -> row M=L, K 0..15;
// lane L>=16 -> row M=L-16, K 16..31.
__device__ __forceinline__ v16bf load_fragA_lds(const unsigned short* buf, int ld, int mbase) {
  int lane = threadIdx.x & 31;
  int m = lane & 15, kb = (lane >> 4) * 16;
  BF16x16 r;
#pragma unroll
  for (int j = 0; j < 8; ++j) {
    r.u[j]     = buf[(mbase + m) * ld + kb + j];
    r.u[8 + j] = buf[(mbase + m) * ld + 8 + kb + j];
  }
  return r.v;
}

// B fragment (32x16, bf16) from LDS: lane L -> col N=L&15; lanes 0-15 hold
// K=0..15, lanes 16-31 hold K=16..31.
__device__ __forceinline__ v16bf load_fragB_lds(const unsigned short* buf, int ld, int nbase) {
  int lane = threadIdx.x & 31;
  int n = lane & 15, kh = (lane >> 4) * 16;
  BF16x16 r;
#pragma unroll
  for (int j = 0; j < 16; ++j)
    r.u[j] = buf[(kh + j) * ld + nbase + n];
  return r.v;
}

// ---------------------------------------------------------------------------
// GEMM: Y[M,N] = A[M,K] @ W[K,N] + bias   (f32 in, bf16 staged, f32 accum)
// 128 threads = 4 waves; block tile 64x64; wave tile 32x32 (2x2 WMMA).
// ---------------------------------------------------------------------------
template <bool OUT_BF16>
__global__ __launch_bounds__(128) void gemm_wmma_kernel(
    const float* __restrict__ A, const float* __restrict__ W,
    const float* __restrict__ bias, void* __restrict__ Yv,
    int M, int N, int K) {
  __shared__ unsigned short sA[64 * 34];
  __shared__ unsigned short sB[32 * 72];

  const int n0 = blockIdx.x * 64, m0 = blockIdx.y * 64;
  const int tid = threadIdx.x, lane = tid & 31, wave = tid >> 5;
  const int wm = (wave >> 1) * 32, wn = (wave & 1) * 32;

  v8f acc[2][2] = {};

  for (int kc = 0; kc < K; kc += 32) {
    // stage A 64x32 (f32 -> bf16)
#pragma unroll
    for (int t = 0; t < 16; ++t) {
      int e = t * 128 + tid; int r = e >> 5, c = e & 31;
      sA[r * 34 + c] = f2bf(A[(size_t)(m0 + r) * K + kc + c]);
    }
    // stage W 32x64 (f32 -> bf16)
#pragma unroll
    for (int t = 0; t < 16; ++t) {
      int e = t * 128 + tid; int r = e >> 6, c = e & 63;
      sB[r * 72 + c] = f2bf(W[(size_t)(kc + r) * N + n0 + c]);
    }
    if (kc + 32 < K) {   // CDNA5 global_prefetch of next K chunk
      __builtin_prefetch(&A[(size_t)(m0 + (tid >> 1)) * K + kc + 32 + (tid & 1) * 16], 0, 1);
      __builtin_prefetch(&W[(size_t)(kc + 32 + (tid >> 2)) * N + n0 + (tid & 3) * 16], 0, 1);
    }
    __syncthreads();

    v16bf a0 = load_fragA_lds(sA, 34, wm);
    v16bf a1 = load_fragA_lds(sA, 34, wm + 16);
    v16bf b0 = load_fragB_lds(sB, 72, wn);
    v16bf b1 = load_fragB_lds(sB, 72, wn + 16);
    acc[0][0] = WMMA_BF16(a0, b0, acc[0][0]);
    acc[0][1] = WMMA_BF16(a0, b1, acc[0][1]);
    acc[1][0] = WMMA_BF16(a1, b0, acc[1][0]);
    acc[1][1] = WMMA_BF16(a1, b1, acc[1][1]);
    __syncthreads();
  }

  // epilogue: C layout -> lane = N (lane&15), M = r + 8*(lane>>4)
  const int half = lane >> 4, nn = lane & 15;
#pragma unroll
  for (int mi = 0; mi < 2; ++mi)
#pragma unroll
    for (int ni = 0; ni < 2; ++ni)
#pragma unroll
      for (int r = 0; r < 8; ++r) {
        int row = m0 + wm + mi * 16 + r + half * 8;
        int col = n0 + wn + ni * 16 + nn;
        float val = acc[mi][ni][r] + bias[col];
        if (OUT_BF16) ((unsigned short*)Yv)[(size_t)row * N + col] = f2bf(val);
        else          ((float*)Yv)[(size_t)row * N + col] = val;
      }
}

// ---------------------------------------------------------------------------
// Banded disentangled attention. 1 wave per (b, h, 16-query tile).
// scores = (q.kT * 8^-1 + q.pkT + pq.kT) * 192^-0.5, band mask, ALiBi,
// softmax, out = P @ V. Writes wei band (f32) + attnout (f32).
// ---------------------------------------------------------------------------
#define T_SEQ 1024
#define NHEAD 16
#define WIN   256
#define CDIM  1024
#define BANDW 288   // 17 key tiles (272) padded to 9x32

__global__ __launch_bounds__(32) void attn_band_kernel(
    const unsigned short* __restrict__ qb,  const unsigned short* __restrict__ kbf,
    const unsigned short* __restrict__ vbf, const unsigned short* __restrict__ pqb,
    const unsigned short* __restrict__ pkb,
    float* __restrict__ wei, float* __restrict__ attnout) {
  __shared__ float          sS[16][BANDW];
  __shared__ unsigned short sP[16][BANDW];
  __shared__ unsigned short sV[32][72];

  const int qi0 = blockIdx.x * 16;
  const int h   = blockIdx.y;
  const int b   = blockIdx.z;
  const int lane = threadIdx.x;
  const int half = lane >> 4, nn = lane & 15;

  const int kbase = (qi0 >= WIN) ? (qi0 - WIN) : 0;
  const int nkt   = (qi0 + 16 - kbase) >> 4;          // <= 17 key tiles
  const size_t base_bh = (size_t)b * T_SEQ * CDIM + h * 64;

  // --- load q / pq A-fragments for this query tile (HEAD=64 -> 2 K-chunks)
  v16bf qA[2], pqA[2];
  {
    const unsigned short* qrow  = qb  + base_bh + (size_t)(qi0 + nn) * CDIM;
    const unsigned short* pqrow = pqb + base_bh + (size_t)(qi0 + nn) * CDIM;
    const int kb8 = half * 16;
#pragma unroll
    for (int c = 0; c < 2; ++c) {
      BF16x16 rq, rp;
#pragma unroll
      for (int j = 0; j < 8; ++j) {
        rq.u[j]     = qrow [c * 32 + kb8 + j];
        rq.u[8 + j] = qrow [c * 32 + 8 + kb8 + j];
        rp.u[j]     = pqrow[c * 32 + kb8 + j];
        rp.u[8 + j] = pqrow[c * 32 + 8 + kb8 + j];
      }
      qA[c] = rq.v; pqA[c] = rp.v;
    }
  }

  const float slope = exp2f(-0.5f * (float)(h + 1));
  const float r64 = 0.125f;                     // 1/sqrt(64)
  const float r192 = 0.07216878364870322f;      // 1/sqrt(192)
  const float NEGINF = -3.0e38f;

  // --- scores: 6 WMMAs per key tile
  for (int kt = 0; kt < nkt; ++kt) {
    const int key0 = kbase + kt * 16;
    const unsigned short* krow  = kbf + base_bh + (size_t)(key0 + nn) * CDIM;
    const unsigned short* pkrow = pkb + base_bh + (size_t)(key0 + nn) * CDIM;
    const int kh = half * 16;
    v16bf kB[2], pkB[2];
#pragma unroll
    for (int c = 0; c < 2; ++c) {
      BF16x16 rk, rp;
#pragma unroll
      for (int j = 0; j < 16; ++j) {
        rk.u[j] = krow [c * 32 + kh + j];   // B[d][n] = k[n][d]: contiguous row
        rp.u[j] = pkrow[c * 32 + kh + j];
      }
      kB[c] = rk.v; pkB[c] = rp.v;
    }
    v8f acc1 = {}, acc2 = {};
    acc1 = WMMA_BF16(qA[0],  kB[0],  acc1);
    acc1 = WMMA_BF16(qA[1],  kB[1],  acc1);
    acc2 = WMMA_BF16(qA[0],  pkB[0], acc2);
    acc2 = WMMA_BF16(qA[1],  pkB[1], acc2);
    acc2 = WMMA_BF16(pqA[0], kB[0],  acc2);
    acc2 = WMMA_BF16(pqA[1], kB[1],  acc2);
#pragma unroll
    for (int r = 0; r < 8; ++r) {
      int m = r + half * 8;
      int i = qi0 + m;
      int j = key0 + nn;
      float raw = (acc1[r] * r64 + acc2[r]) * r192;
      bool ok = (j <= i) && (j >= i - WIN);
      int d = i - j; if (d < 0) d = -d;
      sS[m][kt * 16 + nn] = ok ? (raw - slope * (float)d) : NEGINF;
    }
  }
  // pad unused band columns
  for (int t = nkt * 16 + lane; t < BANDW; t += 32)
    for (int m = 0; m < 16; ++m) sS[m][t] = NEGINF;
  __syncthreads();

  // --- softmax: 2 lanes per row, wave32 xor-16 reduction
  const int row = nn;
  float mx = NEGINF;
  for (int t = half; t < BANDW; t += 2) mx = fmaxf(mx, sS[row][t]);
  mx = fmaxf(mx, __shfl_xor(mx, 16, 32));
  float sum = 0.f;
  for (int t = half; t < BANDW; t += 2) sum += __expf(sS[row][t] - mx);
  sum += __shfl_xor(sum, 16, 32);
  const float inv = 1.0f / sum;
  const int i_row = qi0 + row;
  float* weirow = wei + (((size_t)(b * NHEAD + h)) * T_SEQ + i_row) * T_SEQ;
  for (int t = half; t < BANDW; t += 2) {
    float p = __expf(sS[row][t] - mx) * inv;
    sP[row][t] = f2bf(p);
    if (t < nkt * 16) weirow[kbase + t] = p;   // band only; rest pre-zeroed
  }
  __syncthreads();

  // --- P @ V : 9 chunks of 32 keys, 4 N-tiles of 16 dims
  v8f oacc[4] = {};
  for (int kc = 0; kc < BANDW / 32; ++kc) {
    // Stage V chunk (32 keys x 64 dims, bf16) via the CDNA5 async
    // global->LDS datapath (ASYNCcnt): one contiguous 128B row per lane,
    // issued as 8x global_load_async_to_lds_b128. LDS offset is the low
    // 32 bits of the flat pointer (LDS aperture maps addr[31:0]).
    {
      int key = kbase + kc * 32 + lane;
      int kcl = (key < T_SEQ - 1) ? key : (T_SEQ - 1);  // clamp; P==0 there
      const unsigned short* vrow = vbf + base_bh + (size_t)kcl * CDIM;
      unsigned lds_off = (unsigned)(unsigned long long)(&sV[lane][0]);
#pragma unroll
      for (int d0 = 0; d0 < 64; d0 += 8) {
        unsigned long long ga = (unsigned long long)(vrow + d0);
        unsigned           la = lds_off + (unsigned)(d0 * 2);
        asm volatile("global_load_async_to_lds_b128 %0, %1, off"
                     :: "v"(la), "v"(ga) : "memory");
      }
      asm volatile("s_wait_asynccnt 0" ::: "memory");
    }
    __syncthreads();
    BF16x16 pa;
    const int kb8 = half * 16;
#pragma unroll
    for (int j = 0; j < 8; ++j) {
      pa.u[j]     = sP[nn][kc * 32 + kb8 + j];
      pa.u[8 + j] = sP[nn][kc * 32 + 8 + kb8 + j];
    }
    const int kh = half * 16;
#pragma unroll
    for (int ni = 0; ni < 4; ++ni) {
      BF16x16 vb;
#pragma unroll
      for (int j = 0; j < 16; ++j) vb.u[j] = sV[kh + j][ni * 16 + nn];
      oacc[ni] = WMMA_BF16(pa.v, vb.v, oacc[ni]);
    }
    __syncthreads();
  }

  // --- epilogue: attnout[b,t,C] with head slice at h*64
#pragma unroll
  for (int ni = 0; ni < 4; ++ni)
#pragma unroll
    for (int r = 0; r < 8; ++r) {
      int m = r + half * 8;
      int i2 = qi0 + m;
      int d = ni * 16 + nn;
      attnout[((size_t)(b * T_SEQ + i2)) * CDIM + h * 64 + d] = oacc[ni][r];
    }
}

// ---------------------------------------------------------------------------
extern "C" void kernel_launch(void* const* d_in, const int* in_sizes, int n_in,
                              void* d_out, int out_size, void* d_ws, size_t ws_size,
                              hipStream_t stream) {
  const float* x    = (const float*)d_in[0];
  const float* pos  = (const float*)d_in[1];
  const float* Wq   = (const float*)d_in[2];  const float* bq  = (const float*)d_in[3];
  const float* Wk   = (const float*)d_in[4];  const float* bk  = (const float*)d_in[5];
  const float* Wv   = (const float*)d_in[6];  const float* bv  = (const float*)d_in[7];
  const float* Wpq  = (const float*)d_in[8];  const float* bpq = (const float*)d_in[9];
  const float* Wpk  = (const float*)d_in[10]; const float* bpk = (const float*)d_in[11];
  const float* Wo   = (const float*)d_in[12]; const float* bo  = (const float*)d_in[13];

  const int M = 4 * 1024, N = 1024, K = 1024;
  float* out = (float*)d_out;
  float* wei = out + (size_t)M * N;           // [4,16,1024,1024] f32

  unsigned short* qb   = (unsigned short*)d_ws;
  unsigned short* kbuf = qb   + (size_t)M * N;
  unsigned short* vbuf = kbuf + (size_t)M * N;
  unsigned short* pqb  = vbuf + (size_t)M * N;
  unsigned short* pkb  = pqb  + (size_t)M * N;
  float* attnout = (float*)(pkb + (size_t)M * N);

  // zero wei (masked entries must be exact 0; only the band is written)
  hipMemsetAsync(wei, 0, (size_t)4 * NHEAD * T_SEQ * T_SEQ * sizeof(float), stream);

  dim3 gg(N / 64, M / 64), bb(128);
  gemm_wmma_kernel<true ><<<gg, bb, 0, stream>>>(x,   Wq,  bq,  qb,   M, N, K);
  gemm_wmma_kernel<true ><<<gg, bb, 0, stream>>>(x,   Wk,  bk,  kbuf, M, N, K);
  gemm_wmma_kernel<true ><<<gg, bb, 0, stream>>>(x,   Wv,  bv,  vbuf, M, N, K);
  gemm_wmma_kernel<true ><<<gg, bb, 0, stream>>>(pos, Wpq, bpq, pqb,  M, N, K);
  gemm_wmma_kernel<true ><<<gg, bb, 0, stream>>>(pos, Wpk, bpk, pkb,  M, N, K);

  attn_band_kernel<<<dim3(T_SEQ / 16, NHEAD, 4), dim3(32), 0, stream>>>(
      qb, kbuf, vbuf, pqb, pkb, wei, attnout);

  gemm_wmma_kernel<false><<<gg, bb, 0, stream>>>(attnout, Wo, bo, out, M, N, K);
}
